// ScaledDotProductAttention_67903432950250
// MI455X (gfx1250) — matrix operand order repaired and checked by reading, not compiled, not run on hardware
//
#include <hip/hip_runtime.h>

// ---------------------------------------------------------------------------
// Scaled-dot-product attention for MI455X (gfx1250, wave32, WMMA, async-LDS).
//   attn = softmax(mask ? (q/8)·k^T : -1e9)   [B,H,S,S]  512MB store -> HBM floor
//   out  = v^T·attn·v = v^T·(attn·v)          [B,H,64,64]
// prep : q*0.125 -> f16, k -> f16, v -> f16 transposed per head [d][s]
// pass1: l[row] = sum_t exp(score)            (f16 WMMA, f32 acc, async K tiles)
// pass2: recompute scores, p=exp(x)/l, NT-store attn, W=P·V, W^T -> ws
// pass3: out = V^T·W from global f16 fragments, no LDS, no barriers
// ---------------------------------------------------------------------------

typedef __attribute__((ext_vector_type(16))) _Float16 v16h;
typedef __attribute__((ext_vector_type(8)))  _Float16 v8h;
typedef __attribute__((ext_vector_type(4)))  _Float16 v4h;
typedef __attribute__((ext_vector_type(8)))  float    v8f;

#define Bsz    2
#define Hn     16
#define Ssz    2048
#define Dsz    64
#define BHn    (Bsz*Hn)
#define QSCALE 0.125f
#define NEGBIG (-1.0e9f)

__device__ __forceinline__ v8f wmma_f16(v16h a, v16h b, v8f c) {
  return __builtin_amdgcn_wmma_f32_16x16x32_f16(false, a, false, b, (short)0, c,
                                                false, false);
}

// Low 32 bits of a flat shared-aperture address == workgroup-relative LDS byte
// offset.
__device__ __forceinline__ unsigned lds_lo(const void* p) {
  return (unsigned)(unsigned long long)p;
}
// Async memory->LDS copy, tracked by ASYNCcnt (GVS mode: saddr + 32b voffset).
__device__ __forceinline__ void async_b128(unsigned lds, unsigned voff,
                                           const void* sbase) {
  asm volatile("global_load_async_to_lds_b128 %0, %1, %2"
               :: "v"(lds), "v"(voff), "s"(sbase) : "memory");
}
__device__ __forceinline__ void wait_async_le2() {
  asm volatile("s_wait_asynccnt 0x2" ::: "memory");
}
__device__ __forceinline__ void wait_async_le4() {
  asm volatile("s_wait_asynccnt 0x4" ::: "memory");
}

// ---------------------------------------------------------------------------
// Prep A: q16 = (q/8) as f16, k16 = k as f16 (elementwise, vec4).
// ---------------------------------------------------------------------------
__global__ __launch_bounds__(256)
void sdpa_prep_qk(const float* __restrict__ q, const float* __restrict__ k,
                  _Float16* __restrict__ q16, _Float16* __restrict__ k16) {
  size_t i = (size_t)blockIdx.x * 256 + threadIdx.x;   // i indexes float4
  float4 xq = ((const float4*)q)[i];
  float4 xk = ((const float4*)k)[i];
  v4h hq, hk;
  hq[0] = (_Float16)(xq.x * QSCALE); hq[1] = (_Float16)(xq.y * QSCALE);
  hq[2] = (_Float16)(xq.z * QSCALE); hq[3] = (_Float16)(xq.w * QSCALE);
  hk[0] = (_Float16)xk.x; hk[1] = (_Float16)xk.y;
  hk[2] = (_Float16)xk.z; hk[3] = (_Float16)xk.w;
  *(v4h*)(q16 + i * 4) = hq;
  *(v4h*)(k16 + i * 4) = hk;
}

// ---------------------------------------------------------------------------
// Prep B: vT16[bh][d][s] = (f16)v[bh][s][d]  (64x64 tile transpose via LDS).
// ---------------------------------------------------------------------------
__global__ __launch_bounds__(128)
void sdpa_prep_vT(const float* __restrict__ v, _Float16* __restrict__ vT16) {
  __shared__ __align__(32) _Float16 ldsT[64 * 64];
  const int tid = threadIdx.x;
  const int bh = blockIdx.y, s0 = blockIdx.x * 64;
  const int sl = tid & 63, ch = (tid >> 6) * 32;
  const float* src = v + ((size_t)bh * Ssz + s0 + sl) * Dsz + ch;
#pragma unroll
  for (int j = 0; j < 8; ++j) {
    float4 x = *(const float4*)(src + j * 4);
    ldsT[(ch + j * 4 + 0) * 64 + sl] = (_Float16)x.x;
    ldsT[(ch + j * 4 + 1) * 64 + sl] = (_Float16)x.y;
    ldsT[(ch + j * 4 + 2) * 64 + sl] = (_Float16)x.z;
    ldsT[(ch + j * 4 + 3) * 64 + sl] = (_Float16)x.w;
  }
  __syncthreads();
  const int d = tid >> 1, sh = (tid & 1) * 32;
  _Float16* dst = vT16 + ((size_t)bh * Dsz + d) * Ssz + s0 + sh;
#pragma unroll
  for (int e = 0; e < 4; ++e)
    *(v8h*)(dst + e * 8) = *(const v8h*)&ldsT[d * 64 + sh + e * 8];
}

// Build this wave's Q A-fragments (f16, pre-scaled) from global.
__device__ __forceinline__ void load_q_frags16(const _Float16* __restrict__ qrow,
                                               int g, v16h& aq0, v16h& aq1) {
  v8h a0 = *(const v8h*)&qrow[     16 * g];
  v8h a1 = *(const v8h*)&qrow[ 8 + 16 * g];
  v8h a2 = *(const v8h*)&qrow[32 + 16 * g];
  v8h a3 = *(const v8h*)&qrow[40 + 16 * g];
#pragma unroll
  for (int e = 0; e < 8; ++e) {
    aq0[e] = a0[e]; aq0[8 + e] = a1[e];
    aq1[e] = a2[e]; aq1[8 + e] = a3[e];
  }
}

// ---------------------------------------------------------------------------
// Pass 1: l[row] = sum_t exp(score).  Grid (S/64, B*H), 4 waves/WG.
// Double-buffered async K tiles (32 t-cols x 64 d, f16 = 4KB).
// ---------------------------------------------------------------------------
__global__ __launch_bounds__(128)
void sdpa_pass1_sums(const _Float16* __restrict__ q16,
                     const _Float16* __restrict__ k16,
                     const int* __restrict__ mask,
                     float* __restrict__ l_ws) {
  __shared__ __align__(32) _Float16 ldsK[2][32 * 64];

  const int tid = threadIdx.x;
  const int wave = tid >> 5, lane = tid & 31;
  const int g = lane >> 4, ln = lane & 15;
  const int bh = blockIdx.y, b = bh / Hn;
  const int rowbase = blockIdx.x * 64 + wave * 16;

  v16h aq0, aq1;
  load_q_frags16(q16 + ((size_t)bh * Ssz + rowbase + ln) * Dsz, g, aq0, aq1);

  const _Float16* khead = k16 + (size_t)bh * Ssz * Dsz;
  const int krow = tid >> 2, kcol = (tid & 3) * 16;      // 32B per thread
  const unsigned kvoff = (unsigned)((krow * Dsz + kcol) * 2);
  unsigned kld[2] = { lds_lo(&ldsK[0][krow * Dsz + kcol]),
                      lds_lo(&ldsK[1][krow * Dsz + kcol]) };

  auto issueK = [&](int buf, int t0n) {
    const void* src = khead + (size_t)t0n * Dsz;
    async_b128(kld[buf],      kvoff,      src);
    async_b128(kld[buf] + 16, kvoff + 16, src);
  };

  float srun[8];
#pragma unroll
  for (int r = 0; r < 8; ++r) srun[r] = 0.f;

  issueK(0, 0);
  for (int it = 0; it < Ssz / 32; ++it) {
    const int t0 = it * 32;
    const int tn = (t0 + 32 < Ssz) ? t0 + 32 : 0;     // dummy wrap on last iter
    __syncthreads();                 // everyone done reading buf[(it+1)&1]
    issueK((it + 1) & 1, tn);
    wait_async_le2();                // my tile-it loads landed (in-order)
    __syncthreads();                 // everyone's tile-it loads landed

    const _Float16* Kb = ldsK[it & 1];
    v16h b00 = *(const v16h*)&Kb[ ln       * 64 +      16 * g];
    v16h b01 = *(const v16h*)&Kb[ ln       * 64 + 32 + 16 * g];
    v16h b10 = *(const v16h*)&Kb[(16 + ln) * 64 +      16 * g];
    v16h b11 = *(const v16h*)&Kb[(16 + ln) * 64 + 32 + 16 * g];
    v8f cs0 = {}, cs1 = {};
    cs0 = wmma_f16(aq0, b00, cs0); cs0 = wmma_f16(aq1, b01, cs0);
    cs1 = wmma_f16(aq0, b10, cs1); cs1 = wmma_f16(aq1, b11, cs1);

    const int mv0 = mask[b * Ssz + t0 + ln];
    const int mv1 = mask[b * Ssz + t0 + 16 + ln];
#pragma unroll
    for (int r = 0; r < 8; ++r) {
      srun[r] += __expf(mv0 ? cs0[r] : NEGBIG);
      srun[r] += __expf(mv1 ? cs1[r] : NEGBIG);
    }
  }

#pragma unroll
  for (int r = 0; r < 8; ++r) {
    float s = srun[r];
#pragma unroll
    for (int off = 1; off < 16; off <<= 1) s += __shfl_xor(s, off, 32);
    if (ln == 0) l_ws[(size_t)bh * Ssz + rowbase + r + 8 * g] = s;
  }
}

// ---------------------------------------------------------------------------
// Pass 2: p = exp(score)/l -> NT-store attn; W = P·V; W^T -> ws (coalesced).
// Async double-buffered K tiles and vT tiles.
// ---------------------------------------------------------------------------
__global__ __launch_bounds__(128)
void sdpa_pass2_attn_pv(const _Float16* __restrict__ q16,
                        const _Float16* __restrict__ k16,
                        const _Float16* __restrict__ vT16,
                        const int* __restrict__ mask,
                        const float* __restrict__ l_ws,
                        float* __restrict__ attn,
                        _Float16* __restrict__ wT16) {
  __shared__ __align__(32) _Float16 ldsK[2][32 * 64];
  __shared__ __align__(32) _Float16 ldsVT[2][64 * 32];
  __shared__ __align__(32) _Float16 ldsP[4][16 * 32];
  __shared__ __align__(32) _Float16 ldsWT[64 * 64];

  const int tid = threadIdx.x;
  const int wave = tid >> 5, lane = tid & 31;
  const int g = lane >> 4, ln = lane & 15;
  const int bh = blockIdx.y, b = bh / Hn;
  const int rowb0 = blockIdx.x * 64;
  const int rowbase = rowb0 + wave * 16;

  v16h aq0, aq1;
  load_q_frags16(q16 + ((size_t)bh * Ssz + rowbase + ln) * Dsz, g, aq0, aq1);

  float invl[8];
#pragma unroll
  for (int r = 0; r < 8; ++r)
    invl[r] = 1.0f / l_ws[(size_t)bh * Ssz + rowbase + r + 8 * g];

  const _Float16* khead  = k16  + (size_t)bh * Ssz * Dsz;
  const _Float16* vThead = vT16 + (size_t)bh * Dsz * Ssz;
  const int krow = tid >> 2, kcol = (tid & 3) * 16;
  const unsigned kvoff = (unsigned)((krow * Dsz + kcol) * 2);
  const int vc = tid >> 1, vpart = (tid & 1) * 16;
  const unsigned vvoff = (unsigned)((vc * Ssz + vpart) * 2);
  unsigned kld[2] = { lds_lo(&ldsK[0][krow * Dsz + kcol]),
                      lds_lo(&ldsK[1][krow * Dsz + kcol]) };
  unsigned vld[2] = { lds_lo(&ldsVT[0][vc * 32 + vpart]),
                      lds_lo(&ldsVT[1][vc * 32 + vpart]) };

  auto issueKV = [&](int buf, int t0n) {
    const void* ks = khead + (size_t)t0n * Dsz;
    const void* vs = vThead + t0n;
    async_b128(kld[buf],      kvoff,      ks);
    async_b128(kld[buf] + 16, kvoff + 16, ks);
    async_b128(vld[buf],      vvoff,      vs);
    async_b128(vld[buf] + 16, vvoff + 16, vs);
  };

  v8f accW[4] = {{}, {}, {}, {}};

  issueKV(0, 0);
  for (int it = 0; it < Ssz / 32; ++it) {
    const int t0 = it * 32;
    const int tn = (t0 + 32 < Ssz) ? t0 + 32 : 0;
    __syncthreads();
    issueKV((it + 1) & 1, tn);
    wait_async_le4();
    __syncthreads();

    const _Float16* Kb = ldsK[it & 1];
    const _Float16* Vb = ldsVT[it & 1];
    v16h b00 = *(const v16h*)&Kb[ ln       * 64 +      16 * g];
    v16h b01 = *(const v16h*)&Kb[ ln       * 64 + 32 + 16 * g];
    v16h b10 = *(const v16h*)&Kb[(16 + ln) * 64 +      16 * g];
    v16h b11 = *(const v16h*)&Kb[(16 + ln) * 64 + 32 + 16 * g];
    v8f cs0 = {}, cs1 = {};
    cs0 = wmma_f16(aq0, b00, cs0); cs0 = wmma_f16(aq1, b01, cs0);
    cs1 = wmma_f16(aq0, b10, cs1); cs1 = wmma_f16(aq1, b11, cs1);

    const int mv0 = mask[b * Ssz + t0 + ln];
    const int mv1 = mask[b * Ssz + t0 + 16 + ln];
#pragma unroll
    for (int r = 0; r < 8; ++r) {
      const int mrow = r + 8 * g;
      float p0 = __expf(mv0 ? cs0[r] : NEGBIG) * invl[r];
      float p1 = __expf(mv1 ? cs1[r] : NEGBIG) * invl[r];
      size_t abase = ((size_t)bh * Ssz + rowbase + mrow) * Ssz + t0;
      __builtin_nontemporal_store(p0, attn + abase + ln);        // 512MB stream:
      __builtin_nontemporal_store(p1, attn + abase + 16 + ln);   // keep L2 clean
      ldsP[wave][mrow * 32 +      ln] = (_Float16)p0;
      ldsP[wave][mrow * 32 + 16 + ln] = (_Float16)p1;
    }
    // same-wave LDS ops are in-order (DScnt) -> no barrier needed here
    v8h plo = *(const v8h*)&ldsP[wave][ln * 32 +     16 * g];
    v8h phi = *(const v8h*)&ldsP[wave][ln * 32 + 8 + 16 * g];
    v16h ap;
#pragma unroll
    for (int e = 0; e < 8; ++e) { ap[e] = plo[e]; ap[8 + e] = phi[e]; }
#pragma unroll
    for (int j = 0; j < 4; ++j) {
      v16h bv = *(const v16h*)&Vb[(j * 16 + ln) * 32 + 16 * g];
      accW[j] = wmma_f16(ap, bv, accW[j]);
    }
  }

  // Stage W f16 in LDS, then write W^T rows coalesced (128B per c-row).
  __syncthreads();
#pragma unroll
  for (int j = 0; j < 4; ++j)
#pragma unroll
    for (int r = 0; r < 8; ++r)
      ldsWT[(j * 16 + ln) * 64 + wave * 16 + r + 8 * g] = (_Float16)accW[j][r];
  __syncthreads();
  const int c = tid >> 1, sh = (tid & 1) * 32;
  _Float16* dst = wT16 + ((size_t)bh * Dsz + c) * Ssz + rowb0 + sh;
#pragma unroll
  for (int e = 0; e < 4; ++e)
    *(v8h*)(dst + e * 8) = *(const v8h*)&ldsWT[c * 64 + sh + e * 8];
}

// ---------------------------------------------------------------------------
// Pass 3: out = V^T·W per head from global f16 fragments. 1 WG/head,
// wave w owns d-block [16w,16w+16); no LDS, no barriers.
// ---------------------------------------------------------------------------
__global__ __launch_bounds__(128)
void sdpa_pass3_vtw(const _Float16* __restrict__ vT16,
                    const _Float16* __restrict__ wT16,
                    float* __restrict__ out) {
  const int tid = threadIdx.x;
  const int wave = tid >> 5, lane = tid & 31;
  const int g = lane >> 4, ln = lane & 15;
  const int bh = blockIdx.x;

  const _Float16* vh = vT16 + (size_t)bh * Dsz * Ssz;
  const _Float16* wh = wT16 + (size_t)bh * Dsz * Ssz;
  const int d = wave * 16 + ln;

  v8f acc[4] = {{}, {}, {}, {}};
  for (int s0 = 0; s0 < Ssz; s0 += 32) {
    v8h lo = *(const v8h*)&vh[(size_t)d * Ssz + s0 +     16 * g];
    v8h hi = *(const v8h*)&vh[(size_t)d * Ssz + s0 + 8 + 16 * g];
    v16h av;
#pragma unroll
    for (int e = 0; e < 8; ++e) { av[e] = lo[e]; av[8 + e] = hi[e]; }
#pragma unroll
    for (int j = 0; j < 4; ++j) {
      v16h bw = *(const v16h*)&wh[(size_t)(j * 16 + ln) * Ssz + s0 + 16 * g];
      acc[j] = wmma_f16(av, bw, acc[j]);
    }
  }
#pragma unroll
  for (int j = 0; j < 4; ++j)
#pragma unroll
    for (int r = 0; r < 8; ++r)
      out[(size_t)bh * Dsz * Dsz + (wave * 16 + r + 8 * g) * Dsz + j * 16 + ln]
          = acc[j][r];
}

// ---------------------------------------------------------------------------
extern "C" void kernel_launch(void* const* d_in, const int* in_sizes, int n_in,
                              void* d_out, int out_size, void* d_ws, size_t ws_size,
                              hipStream_t stream) {
  (void)in_sizes; (void)n_in; (void)out_size; (void)ws_size;
  const float* q    = (const float*)d_in[0];
  const float* k    = (const float*)d_in[1];
  const float* v    = (const float*)d_in[2];
  const int*   mask = (const int*)d_in[3];

  float* out  = (float*)d_out;                        // [B,H,64,64]
  float* attn = out + (size_t)BHn * Dsz * Dsz;        // [B,H,S,S]

  const size_t NQK = (size_t)BHn * Ssz * Dsz;         // 4,194,304 elements
  char* w = (char*)d_ws;
  float*    l_ws = (float*)w;                         // BH*S f32     (256KB)
  _Float16* q16  = (_Float16*)(w + ((size_t)BHn * Ssz * 4));
  _Float16* k16  = q16  + NQK;                        // 8MB each
  _Float16* vT16 = k16  + NQK;
  _Float16* wT16 = vT16 + NQK;

  dim3 blk128(128);
  sdpa_prep_qk <<<dim3((unsigned)(NQK / 4 / 256)), dim3(256), 0, stream>>>(q, k, q16, k16);
  sdpa_prep_vT <<<dim3(Ssz / 64, BHn), blk128, 0, stream>>>(v, vT16);
  sdpa_pass1_sums   <<<dim3(Ssz / 64, BHn), blk128, 0, stream>>>(q16, k16, mask, l_ws);
  sdpa_pass2_attn_pv<<<dim3(Ssz / 64, BHn), blk128, 0, stream>>>(q16, k16, vT16, mask,
                                                                 l_ws, attn, wT16);
  sdpa_pass3_vtw    <<<dim3(BHn), blk128, 0, stream>>>(vT16, wT16, out);
}
